// MLA_8358006358172
// MI455X (gfx1250) — compile-verified, matrix operand order checked by reading, not run on hardware
//
#include <hip/hip_runtime.h>
#include <hip/hip_bf16.h>

// ---------------------------------------------------------------------------
// MLA (multi-head latent attention) forward for MI455X / gfx1250.
// All GEMMs + attention run through v_wmma_f32_16x16x32_bf16 (f32 accumulate).
// ---------------------------------------------------------------------------

typedef __attribute__((ext_vector_type(16))) __bf16 v16bf;
typedef __attribute__((ext_vector_type(8)))  float  v8f;

#define D_   2048
#define H_   16
#define HD_  128
#define TQ_  2048
#define BB_  2
#define R_   512

static __device__ __forceinline__ unsigned short f2bf(float f) {
  unsigned int u = __float_as_uint(f);
  u += 0x7FFFu + ((u >> 16) & 1u);          // round-to-nearest-even
  return (unsigned short)(u >> 16);
}

// Load one 16x32 bf16 WMMA operand fragment from LDS (row-major, `stride`
// elements per row). Matches the CDNA5 16-bit A/B layout:
//   lanes 0-15 : K = 0..7 (v0-3), 16..23 (v4-7)
//   lanes16-31 : K = 8..15 (v0-3), 24..31 (v4-7)
// => two 16-byte contiguous LDS reads per lane.
static __device__ __forceinline__ v16bf lds_frag(const unsigned short* p, int stride,
                                                 int row, int koff, int lane) {
  const int kb = (lane & 16) ? 8 : 0;
  const unsigned short* base = p + row * stride + koff + kb;
  union { uint4 u[2]; v16bf v; } c;
  c.u[0] = *(const uint4*)(base);
  c.u[1] = *(const uint4*)(base + 16);
  return c.v;
}

// ---------------------------------------------------------------------------
// fp32 -> bf16 conversion (bandwidth-bound, sizes are multiples of 1024)
// ---------------------------------------------------------------------------
__global__ __launch_bounds__(256) void f32_to_bf16_k(const float* __restrict__ src,
                                                     unsigned short* __restrict__ dst,
                                                     int n) {
  int i = (blockIdx.x * 256 + threadIdx.x) * 4;
  if (i + 3 < n) {
    float4 v = *(const float4*)(src + i);
    dst[i + 0] = f2bf(v.x);
    dst[i + 1] = f2bf(v.y);
    dst[i + 2] = f2bf(v.z);
    dst[i + 3] = f2bf(v.w);
  }
}

// ---------------------------------------------------------------------------
// C[M,N] = A[M,K] (bf16 row-major) x W[N,K]^T (bf16 row-major) (+bias)
// Block: 256 threads (8 waves). Tile: 128(M) x 64(N), K-step 32.
// Wave (wm,wn) computes a 32x32 sub-tile = 2x2 WMMA 16x16 fragments.
// ---------------------------------------------------------------------------
__global__ __launch_bounds__(256) void gemm_abT(const unsigned short* __restrict__ A,
                                                const unsigned short* __restrict__ W,
                                                void* __restrict__ C,
                                                const float* __restrict__ bias,
                                                int M, int N, int K, int out_f32) {
  __shared__ __align__(16) unsigned short As[128 * 40];  // stride 40 keeps b128 alignment
  __shared__ __align__(16) unsigned short Ws[64 * 40];

  const int tid  = threadIdx.x;
  const int lane = tid & 31;
  const int wv   = tid >> 5;
  const int wm   = (wv & 3) * 32;
  const int wn   = (wv >> 2) * 32;
  const int m0   = blockIdx.y * 128;
  const int n0   = blockIdx.x * 64;

  v8f acc[2][2] = {};

  for (int k0 = 0; k0 < K; k0 += 32) {
    // cooperative tile loads: 8 bf16 (16 B) per thread per pass
#pragma unroll
    for (int p = 0; p < 2; ++p) {
      int idx = (p * 256 + tid) * 8;
      int row = idx >> 5, col = idx & 31;
      uint4 v = *(const uint4*)(A + (size_t)(m0 + row) * K + k0 + col);
      *(uint4*)(&As[row * 40 + col]) = v;
    }
    {
      int idx = tid * 8;
      int row = idx >> 5, col = idx & 31;
      uint4 v = *(const uint4*)(W + (size_t)(n0 + row) * K + k0 + col);
      *(uint4*)(&Ws[row * 40 + col]) = v;
    }
    if (k0 + 32 < K) {  // global_prefetch_b8 of the next K tile
      __builtin_prefetch(A + (size_t)(m0 + (tid >> 1)) * K + k0 + 32, 0, 1);
      __builtin_prefetch(W + (size_t)(n0 + (tid & 63)) * K + k0 + 32, 0, 1);
    }
    __syncthreads();

    v16bf af[2], bfm[2];
#pragma unroll
    for (int mi = 0; mi < 2; ++mi)
      af[mi] = lds_frag(As, 40, wm + mi * 16 + (lane & 15), 0, lane);
#pragma unroll
    for (int ni = 0; ni < 2; ++ni)
      bfm[ni] = lds_frag(Ws, 40, wn + ni * 16 + (lane & 15), 0, lane);
#pragma unroll
    for (int mi = 0; mi < 2; ++mi)
#pragma unroll
      for (int ni = 0; ni < 2; ++ni)
        acc[mi][ni] = __builtin_amdgcn_wmma_f32_16x16x32_bf16(
            false, af[mi], false, bfm[ni], (short)0, acc[mi][ni], false, false);
    __syncthreads();
  }

  // Epilogue. C-fragment layout: row = r + 8*(lane>=16), col = lane&15.
  const int rowadd = (lane & 16) ? 8 : 0;
#pragma unroll
  for (int mi = 0; mi < 2; ++mi)
#pragma unroll
    for (int ni = 0; ni < 2; ++ni) {
      const int col = n0 + wn + ni * 16 + (lane & 15);
      const float bv = bias ? bias[col] : 0.f;
#pragma unroll
      for (int r = 0; r < 8; ++r) {
        const int row = m0 + wm + mi * 16 + r + rowadd;
        float v = acc[mi][ni][r] + bv;
        if (out_f32)
          ((float*)C)[(size_t)row * N + col] = v;
        else
          ((unsigned short*)C)[(size_t)row * N + col] = f2bf(v);
      }
    }
}

// ---------------------------------------------------------------------------
// Flash attention per (b,h). Block = 128 threads (4 waves), 64 query rows,
// streaming 64-key tiles. Q/K tiles row-major in LDS, V transposed [hd][key]
// so both GEMMs use contiguous b128 LDS fragment reads.
// ---------------------------------------------------------------------------
__global__ __launch_bounds__(128) void mla_attn(const unsigned short* __restrict__ Q,
                                                const unsigned short* __restrict__ Kx,
                                                const unsigned short* __restrict__ V,
                                                unsigned short* __restrict__ O) {
  __shared__ __align__(16) unsigned short Qs[64 * 136];
  __shared__ __align__(16) unsigned short Ks[64 * 136];
  __shared__ __align__(16) unsigned short Vs[128 * 72];      // transposed: [hd][key]
  __shared__ __align__(16) unsigned short Ps[4 * 16 * 72];   // per-wave P staging

  const int tid  = threadIdx.x;
  const int lane = tid & 31;
  const int wv   = tid >> 5;
  const int q0   = blockIdx.x * 64;
  const int bh   = blockIdx.y;
  const int b    = bh / H_;
  const int h    = bh % H_;
  const size_t rs   = (size_t)H_ * HD_;                      // 2048 row stride
  const size_t base = (size_t)b * TQ_ * rs + (size_t)h * HD_;

  // Q tile 64x128, loaded once
#pragma unroll
  for (int p = 0; p < 8; ++p) {
    int idx = (p * 128 + tid) * 8;
    int row = idx >> 7, col = idx & 127;
    uint4 v = *(const uint4*)(Q + base + (size_t)(q0 + row) * rs + col);
    *(uint4*)(&Qs[row * 136 + col]) = v;
  }

  v8f o[8] = {};
  float mrow[8], lrow[8];
#pragma unroll
  for (int r = 0; r < 8; ++r) { mrow[r] = -1e30f; lrow[r] = 0.f; }

  unsigned short* Pw  = &Ps[wv * 16 * 72];
  const int rowadd    = (lane & 16) ? 8 : 0;
  const float sc      = 0.08838834764831845f;                // 1/sqrt(128)

  for (int kb = 0; kb < TQ_ / 64; ++kb) {
    const int key0 = kb * 64;
    __syncthreads();  // previous PV readers done before K/V overwrite
#pragma unroll
    for (int p = 0; p < 8; ++p) {
      int idx = (p * 128 + tid) * 8;
      int row = idx >> 7, col = idx & 127;
      uint4 v = *(const uint4*)(Kx + base + (size_t)(key0 + row) * rs + col);
      *(uint4*)(&Ks[row * 136 + col]) = v;
    }
#pragma unroll
    for (int p = 0; p < 8; ++p) {
      int idx = (p * 128 + tid) * 8;
      int row = idx >> 7, col = idx & 127;
      uint4 v = *(const uint4*)(V + base + (size_t)(key0 + row) * rs + col);
      const unsigned short* e = (const unsigned short*)&v;
#pragma unroll
      for (int j = 0; j < 8; ++j) Vs[(col + j) * 72 + row] = e[j];  // transpose
    }
    __syncthreads();

    // S(16x64) = Q(16x128) x K^T : B-frag col = key -> row of Ks tile
    v8f s[4] = {};
#pragma unroll
    for (int ks = 0; ks < 4; ++ks) {
      v16bf aq = lds_frag(Qs, 136, wv * 16 + (lane & 15), ks * 32, lane);
#pragma unroll
      for (int n = 0; n < 4; ++n) {
        v16bf bk = lds_frag(Ks, 136, n * 16 + (lane & 15), ks * 32, lane);
        s[n] = __builtin_amdgcn_wmma_f32_16x16x32_bf16(false, aq, false, bk,
                                                       (short)0, s[n], false, false);
      }
    }

    // Online softmax directly in C-fragment layout: row r lives in 16 lanes
    // (same half-wave), cols spread over the 4 n-tiles.
    float corr[8];
#pragma unroll
    for (int r = 0; r < 8; ++r) {
      float v = -1e30f;
#pragma unroll
      for (int n = 0; n < 4; ++n) { s[n][r] *= sc; v = fmaxf(v, s[n][r]); }
#pragma unroll
      for (int m = 1; m < 16; m <<= 1) v = fmaxf(v, __shfl_xor(v, m, 32));
      const float mn = fmaxf(mrow[r], v);
      corr[r] = __expf(mrow[r] - mn);
      mrow[r] = mn;
      float psum = 0.f;
#pragma unroll
      for (int n = 0; n < 4; ++n) {
        float p = __expf(s[n][r] - mn);
        s[n][r] = p;
        psum += p;
      }
#pragma unroll
      for (int m = 1; m < 16; m <<= 1) psum += __shfl_xor(psum, m, 32);
      lrow[r] = lrow[r] * corr[r] + psum;
    }

    // Stage P as bf16 (C-layout -> row-major) and rescale O accumulators
#pragma unroll
    for (int n = 0; n < 4; ++n)
#pragma unroll
      for (int r = 0; r < 8; ++r)
        Pw[(r + rowadd) * 72 + n * 16 + (lane & 15)] = f2bf(s[n][r]);
#pragma unroll
    for (int n8 = 0; n8 < 8; ++n8)
#pragma unroll
      for (int r = 0; r < 8; ++r) o[n8][r] *= corr[r];
    __syncthreads();

    // O(16x128) += P(16x64) x V(64x128): B-frag col = hd -> row of Vs (transposed)
#pragma unroll
    for (int kk = 0; kk < 2; ++kk) {
      v16bf ap = lds_frag(Pw, 72, lane & 15, kk * 32, lane);
#pragma unroll
      for (int n8 = 0; n8 < 8; ++n8) {
        v16bf bv = lds_frag(Vs, 72, n8 * 16 + (lane & 15), kk * 32, lane);
        o[n8] = __builtin_amdgcn_wmma_f32_16x16x32_bf16(false, ap, false, bv,
                                                        (short)0, o[n8], false, false);
      }
    }
  }

  // Normalize and store (bf16, feeds the output projection GEMM)
#pragma unroll
  for (int n8 = 0; n8 < 8; ++n8) {
    const int hd = n8 * 16 + (lane & 15);
#pragma unroll
    for (int r = 0; r < 8; ++r) {
      const int t = q0 + wv * 16 + r + rowadd;
      O[base + (size_t)t * rs + hd] = f2bf(o[n8][r] / lrow[r]);
    }
  }
}

// ---------------------------------------------------------------------------
// Launch chain
// ---------------------------------------------------------------------------
extern "C" void kernel_launch(void* const* d_in, const int* in_sizes, int n_in,
                              void* d_out, int out_size, void* d_ws, size_t ws_size,
                              hipStream_t stream) {
  (void)in_sizes; (void)n_in; (void)out_size; (void)ws_size;

  const float* x    = (const float*)d_in[0];
  const float* wq_a = (const float*)d_in[1];
  const float* wq_b = (const float*)d_in[2];
  const float* wk_a = (const float*)d_in[3];
  const float* wk_b = (const float*)d_in[4];
  const float* wv_a = (const float*)d_in[5];
  const float* wv_b = (const float*)d_in[6];
  const float* wo_w = (const float*)d_in[7];
  const float* wo_b = (const float*)d_in[8];

  const int M = BB_ * TQ_;  // 4096 tokens

  size_t off = 0;
  auto alloc = [&](size_t elems) {
    unsigned short* p = (unsigned short*)((char*)d_ws + off);
    off += (elems * sizeof(unsigned short) + 255) & ~(size_t)255;
    return p;
  };
  unsigned short* xb   = alloc((size_t)M * D_);
  unsigned short* wqab = alloc((size_t)R_ * D_);
  unsigned short* wqbb = alloc((size_t)D_ * R_);
  unsigned short* wkab = alloc((size_t)R_ * D_);
  unsigned short* wkbb = alloc((size_t)D_ * R_);
  unsigned short* wvab = alloc((size_t)R_ * D_);
  unsigned short* wvbb = alloc((size_t)D_ * R_);
  unsigned short* wob  = alloc((size_t)D_ * D_);
  unsigned short* cq   = alloc((size_t)M * R_);
  unsigned short* ck   = alloc((size_t)M * R_);
  unsigned short* cv   = alloc((size_t)M * R_);
  unsigned short* qb   = alloc((size_t)M * D_);
  unsigned short* kb   = alloc((size_t)M * D_);
  unsigned short* vb   = alloc((size_t)M * D_);
  unsigned short* ao   = alloc((size_t)M * D_);

  auto conv = [&](const float* s, unsigned short* d, size_t n) {
    f32_to_bf16_k<<<dim3((unsigned)(n / 1024)), dim3(256), 0, stream>>>(s, d, (int)n);
  };
  conv(x,    xb,   (size_t)M * D_);
  conv(wq_a, wqab, (size_t)R_ * D_);
  conv(wq_b, wqbb, (size_t)D_ * R_);
  conv(wk_a, wkab, (size_t)R_ * D_);
  conv(wk_b, wkbb, (size_t)D_ * R_);
  conv(wv_a, wvab, (size_t)R_ * D_);
  conv(wv_b, wvbb, (size_t)D_ * R_);
  conv(wo_w, wob,  (size_t)D_ * D_);

  auto gemm = [&](const unsigned short* A, const unsigned short* W, void* C,
                  const float* bias, int Mg, int Ng, int Kg, int f32out) {
    gemm_abT<<<dim3(Ng / 64, Mg / 128), dim3(256), 0, stream>>>(A, W, C, bias,
                                                                Mg, Ng, Kg, f32out);
  };

  // latent down-projections: [4096,2048] x [512,2048]^T
  gemm(xb, wqab, cq, nullptr, M, R_, D_, 0);
  gemm(xb, wkab, ck, nullptr, M, R_, D_, 0);
  gemm(xb, wvab, cv, nullptr, M, R_, D_, 0);
  // up-projections: [4096,512] x [2048,512]^T
  gemm(cq, wqbb, qb, nullptr, M, D_, R_, 0);
  gemm(ck, wkbb, kb, nullptr, M, D_, R_, 0);
  gemm(cv, wvbb, vb, nullptr, M, D_, R_, 0);
  // flash attention: grid = (T/64 q-blocks, B*H heads)
  mla_attn<<<dim3(TQ_ / 64, BB_ * H_), dim3(128), 0, stream>>>(qb, kb, vb, ao);
  // output projection + bias -> f32 d_out
  gemm(ao, wob, (float*)d_out, wo_b, M, D_, D_, 1);
}
